// PointNet2MSGSeg_32263794328110
// MI455X (gfx1250) — matrix operand order repaired and checked by reading, not compiled, not run on hardware
//
#include <hip/hip_runtime.h>
#include <hip/hip_bf16.h>

typedef __attribute__((ext_vector_type(16))) _Float16 v16h;
typedef __attribute__((ext_vector_type(8)))  _Float16 v8h;
typedef __attribute__((ext_vector_type(4)))  _Float16 v4h;
typedef __attribute__((ext_vector_type(8)))  float    v8f;

#define CDIV(a, b) (((a) + (b) - 1) / (b))

#define TM 64
#define TN 64
#define TK 32

// ============================================================================
// Fused GEMM:  C[M,N] = act( A[M,Kp] @ Wh[Kp,N] * scale[N] + bias[N] )
//
// Preconditions (guaranteed by host orchestration):
//   - M % 64 == 0, lda % 32 == 0; A rows zero-padded beyond logical K
//   - Wh is f16, transposed+padded: Wh[n*ldw + k], n < roundup(N,64),
//     k < ldw == lda, zero-filled outside (K,N)
//   - epilogue writes cols [0, ldc): real cols get scale/bias/relu, pad cols
//     get exact 0 so the next layer's padded-K A input is already valid
//
// 64x64 macro tile / 128-thread block (4 waves); wave = 16-row strip, four
// 16x16 accumulators -> 4 v_wmma_f32_16x16x32_f16 per K-step. A staged in LDS
// as f16 (aligned float4 global loads, v4h LDS stores); A fragments are two
// aligned ds_load_b128. B fragments load directly from global Wh (two aligned
// b128 per tile) -- W is small and hot in WGP$/L2. No per-element predication
// anywhere in the K loop.
// ============================================================================
__global__ __launch_bounds__(128) void wmma_gemm_kernel(
    const float* __restrict__ A, int lda,
    const _Float16* __restrict__ Wh, int ldw,
    const float* __restrict__ scale, const float* __restrict__ bias,
    float* __restrict__ C, int ldc, int M, int N, int do_relu)
{
    __shared__ _Float16 As[TM][TK];    // 4 KB

    const int tid  = threadIdx.x;      // 0..127
    const int l    = tid & 31;
    const int wave = tid >> 5;         // 0..3
    const int half = (l >> 4) & 1;
    const int m16  = l & 15;

    const int row0 = blockIdx.x * TM;
    const int col0 = blockIdx.y * TN;

    const _Float16* wbase = Wh + (size_t)col0 * ldw;

    v8f acc[4] = {{}, {}, {}, {}};

    for (int k0 = 0; k0 < lda; k0 += TK) {
        // ---- stage A tile: 64x32 f32 -> f16, aligned float4 loads ----
        #pragma unroll
        for (int i = 0; i < 4; ++i) {
            int e  = tid + 128 * i;            // 0..511
            int r  = e >> 3;                   // 0..63
            int k4 = (e & 7) * 4;
            const float4 f =
                *(const float4*)(A + (size_t)(row0 + r) * lda + (k0 + k4));
            v4h h = {(_Float16)f.x, (_Float16)f.y, (_Float16)f.z, (_Float16)f.w};
            *(v4h*)&As[r][k4] = h;
        }
        __syncthreads();

        // ---- A fragment: two aligned b128 LDS loads ----
        union V16 { v16h v; v8h h[2]; };
        V16 a;
        const int ar = wave * 16 + m16;
        const int kb = half * 8;
        a.h[0] = *(const v8h*)&As[ar][kb];
        a.h[1] = *(const v8h*)&As[ar][kb + 16];

        // ---- B fragments straight from global Wh; 4 WMMAs ----
        #pragma unroll
        for (int ct = 0; ct < 4; ++ct) {
            V16 b;
            const _Float16* wp =
                wbase + (size_t)(ct * 16 + m16) * ldw + (k0 + half * 8);
            b.h[0] = *(const v8h*)wp;
            b.h[1] = *(const v8h*)(wp + 16);
            acc[ct] = __builtin_amdgcn_wmma_f32_16x16x32_f16(
                /*neg_a=*/false, a.v, /*neg_b=*/false, b.v,
                /*c_mod=*/(short)0, acc[ct], /*reuse_a=*/false, /*reuse_b=*/false);
        }
        __syncthreads();
    }

    // ---- epilogue: branchless pad-fill; C/D layout M = r + 8*half ----
    const int orow = row0 + wave * 16 + half * 8;
    #pragma unroll
    for (int ct = 0; ct < 4; ++ct) {
        int col = col0 + ct * 16 + m16;
        if (col >= ldc) continue;
        bool valid = (col < N);
        float sc = valid ? (scale ? scale[col] : 1.0f) : 0.0f;
        float bs = (valid && bias) ? bias[col] : 0.0f;
        #pragma unroll
        for (int r = 0; r < 8; ++r) {
            float v = acc[ct][r] * sc + bs;
            if (do_relu) v = fmaxf(v, 0.0f);
            C[(size_t)(orow + r) * ldc + col] = v;
        }
    }
}

// ============================================================================
// Per-layer weight prep: W (f32, K x N) -> Wh (f16, [Np64][Kp], zero-padded)
// ============================================================================
__global__ void convert_w_kernel(const float* __restrict__ W,
                                 _Float16* __restrict__ Wh,
                                 int K, int N, int Kp, int total)
{
    int t = blockIdx.x * blockDim.x + threadIdx.x;
    if (t >= total) return;                    // total = Np64*Kp
    int k = t % Kp;
    int n = t / Kp;
    _Float16 v = (_Float16)0.0f;
    if (k < K && n < N) v = (_Float16)W[(size_t)k * N + n];
    Wh[t] = v;
}

// ============================================================================
// Split pointcloud (B,N,9) -> xyz (B,N,3) and feats (B,N,6)
// ============================================================================
__global__ void split_pc_kernel(const float* __restrict__ pc,
                                float* __restrict__ xyz, float* __restrict__ f0,
                                int total)
{
    int t = blockIdx.x * blockDim.x + threadIdx.x;
    if (t >= total) return;
    const float* p = pc + (size_t)t * 9;
    xyz[t * 3 + 0] = p[0];
    xyz[t * 3 + 1] = p[1];
    xyz[t * 3 + 2] = p[2];
    #pragma unroll
    for (int c = 0; c < 6; ++c) f0[(size_t)t * 6 + c] = p[3 + c];
}

// ============================================================================
// Farthest point sampling. One block per batch; distances live in LDS.
// ============================================================================
__global__ __launch_bounds__(256) void fps_kernel(
    const float* __restrict__ xyz, int* __restrict__ fidx, int n, int np)
{
    __shared__ float dists[4096];
    __shared__ float red_v[256];
    __shared__ int   red_i[256];

    const int b = blockIdx.x;
    const int t = threadIdx.x;
    const int T = blockDim.x;
    const float* p = xyz + (size_t)b * n * 3;

    for (int i = t; i < n; i += T) dists[i] = 1e10f;
    __syncthreads();

    int far = 0;
    for (int s = 0; s < np; ++s) {
        if (t == 0) fidx[b * np + s] = far;
        float fx = p[far * 3 + 0], fy = p[far * 3 + 1], fz = p[far * 3 + 2];
        float best = -1.0f; int bi = n;
        for (int i = t; i < n; i += T) {
            float dx = p[i * 3 + 0] - fx;
            float dy = p[i * 3 + 1] - fy;
            float dz = p[i * 3 + 2] - fz;
            float d  = dx * dx + dy * dy + dz * dz;
            float dd = fminf(dists[i], d);
            dists[i] = dd;
            if (dd > best) { best = dd; bi = i; }
        }
        red_v[t] = best; red_i[t] = bi;
        __syncthreads();
        for (int off = T >> 1; off > 0; off >>= 1) {
            if (t < off) {
                float ov = red_v[t + off]; int oi = red_i[t + off];
                if (ov > red_v[t] || (ov == red_v[t] && oi < red_i[t])) {
                    red_v[t] = ov; red_i[t] = oi;
                }
            }
            __syncthreads();
        }
        far = red_i[0];
        __syncthreads();
    }
}

// ============================================================================
// Gather sampled centers: new_xyz[b,s,:] = xyz[b, fidx[b,s], :]
// ============================================================================
__global__ void gather_xyz_kernel(const float* __restrict__ xyz,
                                  const int* __restrict__ fidx,
                                  float* __restrict__ new_xyz,
                                  int n, int np, int total)
{
    int t = blockIdx.x * blockDim.x + threadIdx.x;
    if (t >= total) return;                      // total = B*np*3
    int d  = t % 3;
    int s  = (t / 3) % np;
    int b  = t / (3 * np);
    int j  = fidx[b * np + s];
    new_xyz[t] = xyz[((size_t)b * n + j) * 3 + d];
}

// ============================================================================
// Ball query: first `ns` indices (ascending) with sqdist <= r^2, pad w/ first.
// ============================================================================
__global__ void ballquery_kernel(const float* __restrict__ xyz,
                                 const float* __restrict__ new_xyz,
                                 int* __restrict__ gidx,
                                 int n, int np, float r2, int ns, int total)
{
    int t = blockIdx.x * blockDim.x + threadIdx.x;
    if (t >= total) return;                      // total = B*np
    int b = t / np;
    const float* p = xyz + (size_t)b * n * 3;
    float qx = new_xyz[t * 3 + 0];
    float qy = new_xyz[t * 3 + 1];
    float qz = new_xyz[t * 3 + 2];
    int* out = gidx + (size_t)t * ns;
    int cnt = 0, first = 0;
    for (int j = 0; j < n && cnt < ns; ++j) {
        float dx = p[j * 3 + 0] - qx;
        float dy = p[j * 3 + 1] - qy;
        float dz = p[j * 3 + 2] - qz;
        float d  = dx * dx + dy * dy + dz * dz;
        if (!(d > r2)) {
            if (cnt == 0) first = j;
            out[cnt++] = j;
        }
    }
    for (; cnt < ns; ++cnt) out[cnt] = first;
}

// ============================================================================
// Grouped MLP input rows, K-padded: [gxyz-center | feats | 0...], width Kp
// ============================================================================
__global__ void group_kernel(const float* __restrict__ xyz,
                             const float* __restrict__ feats,
                             const float* __restrict__ new_xyz,
                             const int* __restrict__ gidx,
                             float* __restrict__ out,
                             int n, int np, int ns, int cin, int Kp,
                             long long total)
{
    long long stride = (long long)gridDim.x * blockDim.x;
    for (long long e = blockIdx.x * (long long)blockDim.x + threadIdx.x;
         e < total; e += stride) {
        int c    = (int)(e % Kp);
        long long row = e / Kp;
        int sp   = (int)((row / ns) % np);
        int b    = (int)(row / ((long long)ns * np));
        float v = 0.0f;
        if (c < 3 + cin) {
            int j = gidx[row];
            if (c < 3) {
                v = xyz[((size_t)b * n + j) * 3 + c]
                  - new_xyz[((size_t)b * np + sp) * 3 + c];
            } else {
                v = feats[((size_t)b * n + j) * cin + (c - 3)];
            }
        }
        out[e] = v;
    }
}

// ============================================================================
// Max-pool over the ns samples; write into concatenated feature buffer.
// ============================================================================
__global__ void maxpool_kernel(const float* __restrict__ in,
                               float* __restrict__ out,
                               int rows, int ns, int C, int Ctot, int coff,
                               int total)
{
    int t = blockIdx.x * blockDim.x + threadIdx.x;
    if (t >= total) return;                      // total = rows*C
    int c   = t % C;
    int row = t / C;
    float m = -3.0e38f;
    for (int k = 0; k < ns; ++k)
        m = fmaxf(m, in[((size_t)row * ns + k) * C + c]);
    out[(size_t)row * Ctot + coff + c] = m;
}

// ============================================================================
// 3-NN with inverse-distance weights (normalized), matching top_k(-sqd, 3).
// ============================================================================
__global__ void threenn_kernel(const float* __restrict__ q,
                               const float* __restrict__ s,
                               int* __restrict__ idx3, float* __restrict__ w3,
                               int nq, int nsrc, int total)
{
    int t = blockIdx.x * blockDim.x + threadIdx.x;
    if (t >= total) return;                      // total = B*nq
    int b = t / nq;
    float qx = q[t * 3 + 0], qy = q[t * 3 + 1], qz = q[t * 3 + 2];
    const float* sp = s + (size_t)b * nsrc * 3;
    float d0 = 1e30f, d1 = 1e30f, d2 = 1e30f;
    int   i0 = 0, i1 = 0, i2 = 0;
    for (int j = 0; j < nsrc; ++j) {
        float dx = sp[j * 3 + 0] - qx;
        float dy = sp[j * 3 + 1] - qy;
        float dz = sp[j * 3 + 2] - qz;
        float d  = dx * dx + dy * dy + dz * dz;
        if (d < d0)      { d2 = d1; i2 = i1; d1 = d0; i1 = i0; d0 = d; i0 = j; }
        else if (d < d1) { d2 = d1; i2 = i1; d1 = d;  i1 = j; }
        else if (d < d2) { d2 = d;  i2 = j; }
    }
    float w0 = 1.0f / (fmaxf(d0, 0.0f) + 1e-8f);
    float w1 = 1.0f / (fmaxf(d1, 0.0f) + 1e-8f);
    float w2 = 1.0f / (fmaxf(d2, 0.0f) + 1e-8f);
    float ws = w0 + w1 + w2;
    w0 /= ws; w1 /= ws; w2 /= ws;
    idx3[t * 3 + 0] = i0; idx3[t * 3 + 1] = i1; idx3[t * 3 + 2] = i2;
    w3[t * 3 + 0]  = w0; w3[t * 3 + 1]  = w1; w3[t * 3 + 2]  = w2;
}

// ============================================================================
// Interp (Cs) + skip (Ck) concat, K-padded to width Kp (zeros beyond Cs+Ck).
// ============================================================================
__global__ void interp_concat_kernel(const float* __restrict__ fsrc,
                                     const float* __restrict__ fskip,
                                     const int* __restrict__ idx3,
                                     const float* __restrict__ w3,
                                     float* __restrict__ out,
                                     int nq, int nsrc, int Cs, int Ck, int Kp,
                                     long long total)
{
    long long stride = (long long)gridDim.x * blockDim.x;
    for (long long e = blockIdx.x * (long long)blockDim.x + threadIdx.x;
         e < total; e += stride) {
        int c   = (int)(e % Kp);
        int row = (int)(e / Kp);
        int b   = row / nq;
        float v = 0.0f;
        if (c < Cs) {
            #pragma unroll
            for (int j = 0; j < 3; ++j) {
                int sj = idx3[row * 3 + j];
                v += w3[row * 3 + j] * fsrc[((size_t)b * nsrc + sj) * Cs + c];
            }
        } else if (c < Cs + Ck) {
            v = fskip[(size_t)row * Ck + (c - Cs)];
        }
        out[e] = v;
    }
}

// ============================================================================
// (B,N,ld) -> (B,K,N) taking only first kcls of each padded row
// ============================================================================
__global__ void transpose_out_kernel(const float* __restrict__ in,
                                     float* __restrict__ out,
                                     int n, int kcls, int ld, int total)
{
    int t = blockIdx.x * blockDim.x + threadIdx.x;
    if (t >= total) return;                      // total = B*kcls*n
    int nn = t % n;
    int k  = (t / n) % kcls;
    int b  = t / (n * kcls);
    out[t] = in[((size_t)b * n + nn) * ld + k];
}

// ============================================================================
// Host orchestration
// ============================================================================
static inline size_t align256(size_t x) { return (x + 255) & ~(size_t)255; }
static inline int rup(int x, int m) { return ((x + m - 1) / m) * m; }

extern "C" void kernel_launch(void* const* d_in, const int* in_sizes, int n_in,
                              void* d_out, int out_size, void* d_ws, size_t ws_size,
                              hipStream_t stream)
{
    (void)in_sizes; (void)n_in; (void)out_size;

    const int B = 8, N0 = 4096;
    const int NP[5]      = {4096, 1024, 256, 64, 16};
    const float RAD[4][2] = {{0.05f, 0.1f}, {0.1f, 0.2f}, {0.2f, 0.4f}, {0.4f, 0.8f}};
    const int NSAMP[2]   = {16, 32};
    const int SA_DIMS[4][2][4] = {
        {{9, 16, 16, 32},    {9, 32, 32, 64}},
        {{99, 64, 64, 128},  {99, 64, 96, 128}},
        {{259, 128, 196, 256}, {259, 128, 196, 256}},
        {{515, 256, 256, 512}, {515, 256, 384, 512}}};
    const int FP_DIMS[4][3] = {
        {262, 128, 128}, {608, 256, 256}, {768, 512, 512}, {1536, 512, 512}};

    // ---- parse inputs in setup_inputs() order ----
    int cur = 0;
    const float* pc = (const float*)d_in[cur++];
    const float *saW[4][2][3], *saS[4][2][3], *saB_[4][2][3];
    for (int lvl = 0; lvl < 4; ++lvl)
        for (int s = 0; s < 2; ++s)
            for (int L = 0; L < 3; ++L) {
                saW[lvl][s][L]  = (const float*)d_in[cur++];
                saS[lvl][s][L]  = (const float*)d_in[cur++];
                saB_[lvl][s][L] = (const float*)d_in[cur++];
            }
    const float *fpW[4][2], *fpS[4][2], *fpB[4][2];
    for (int lvl = 0; lvl < 4; ++lvl)
        for (int L = 0; L < 2; ++L) {
            fpW[lvl][L] = (const float*)d_in[cur++];
            fpS[lvl][L] = (const float*)d_in[cur++];
            fpB[lvl][L] = (const float*)d_in[cur++];
        }
    const float* fc1W = (const float*)d_in[cur++];
    const float* fc1S = (const float*)d_in[cur++];
    const float* fc1B = (const float*)d_in[cur++];
    const float* fc2W = (const float*)d_in[cur++];
    const float* fc2B = (const float*)d_in[cur++];

    // ---- carve workspace ----
    char*  ws  = (char*)d_ws;
    size_t off = 0;
    auto alloc = [&](size_t bytes) -> void* {
        void* p = ws + off;
        off = align256(off + bytes);
        return p;
    };
    float* xyz[5];
    for (int l = 0; l < 5; ++l) xyz[l] = (float*)alloc((size_t)B * NP[l] * 3 * 4);
    const size_t FEAT_CAP[5] = {(size_t)B * 4096 * 128, (size_t)B * 1024 * 256,
                                (size_t)B * 256 * 512,  (size_t)B * 64 * 512,
                                (size_t)B * 16 * 1024};
    float* feat[5];
    for (int l = 0; l < 5; ++l) feat[l] = (float*)alloc(FEAT_CAP[l] * 4);
    int*   fidx = (int*)alloc((size_t)B * 1024 * 4);
    int*   gidx = (int*)alloc((size_t)B * 1024 * 32 * 4);
    int*   idx3 = (int*)alloc((size_t)B * 4096 * 3 * 4);
    float* w3   = (float*)alloc((size_t)B * 4096 * 3 * 4);
    const size_t SCR = (size_t)10 * 1024 * 1024;  // floats
    float* scrA = (float*)alloc(SCR * 4);
    float* scrB = (float*)alloc(SCR * 4);
    _Float16* WhBuf = (_Float16*)alloc((size_t)(1 << 20) * 2);  // 1M f16
    if (off > ws_size) return;  // insufficient workspace; nothing sane to do

    const int TPB = 256;

    // Fused layer: convert W to f16 padded-transposed, then WMMA GEMM.
    // Returns ldc (row stride of output, = roundup(N,32)).
    auto gemm = [&](int rows, const float* Ain, int lda, const float* Wf,
                    int K, int N, const float* sc, const float* bs,
                    float* Cout, int relu) -> int {
        int Kp   = rup(K, 32);     // == lda by construction
        int Np64 = rup(N, 64);
        int ldc  = rup(N, 32);
        {
            int total = Np64 * Kp;
            convert_w_kernel<<<CDIV(total, TPB), TPB, 0, stream>>>(
                Wf, WhBuf, K, N, Kp, total);
        }
        dim3 g(rows / TM, CDIV(ldc, TN));
        wmma_gemm_kernel<<<g, 128, 0, stream>>>(
            Ain, lda, WhBuf, Kp, sc, bs, Cout, ldc, rows, N, relu);
        return ldc;
    };

    // ---- stage 0: split pointcloud ----
    {
        int total = B * N0;
        split_pc_kernel<<<CDIV(total, TPB), TPB, 0, stream>>>(pc, xyz[0], feat[0], total);
    }

    int Ccur[5];
    Ccur[0] = 6;

    // ---- set-abstraction levels ----
    for (int lvl = 0; lvl < 4; ++lvl) {
        int nin = NP[lvl], np = NP[lvl + 1];
        fps_kernel<<<B, 256, 0, stream>>>(xyz[lvl], fidx, nin, np);
        {
            int total = B * np * 3;
            gather_xyz_kernel<<<CDIV(total, TPB), TPB, 0, stream>>>(
                xyz[lvl], fidx, xyz[lvl + 1], nin, np, total);
        }
        int coff = 0;
        for (int s = 0; s < 2; ++s) {
            int   ns = NSAMP[s];
            float r  = RAD[lvl][s];
            {
                int total = B * np;
                ballquery_kernel<<<CDIV(total, TPB), TPB, 0, stream>>>(
                    xyz[lvl], xyz[lvl + 1], gidx, nin, np, r * r, ns, total);
            }
            int cin  = Ccur[lvl];
            int rows = B * np * ns;
            int Kp   = rup(3 + cin, 32);
            {
                long long total = (long long)rows * Kp;
                group_kernel<<<CDIV(total, TPB), TPB, 0, stream>>>(
                    xyz[lvl], feat[lvl], xyz[lvl + 1], gidx, scrA,
                    nin, np, ns, cin, Kp, total);
            }
            // 3-layer MLP: scrA -> scrB -> scrA -> scrB (K-padded chain)
            const float* in   = scrA;
            float*       outb = scrB;
            int          kdim = 3 + cin;
            int          lda  = Kp;
            for (int L = 0; L < 3; ++L) {
                int ndim = SA_DIMS[lvl][s][L + 1];
                outb = (L & 1) ? scrA : scrB;
                lda = gemm(rows, in, lda, saW[lvl][s][L], kdim, ndim,
                           saS[lvl][s][L], saB_[lvl][s][L], outb, 1);
                in = outb; kdim = ndim;
            }
            int cout = SA_DIMS[lvl][s][3];   // mult of 32 -> ldc == cout
            int Ctot = SA_DIMS[lvl][0][3] + SA_DIMS[lvl][1][3];
            {
                int total = B * np * cout;
                maxpool_kernel<<<CDIV(total, TPB), TPB, 0, stream>>>(
                    outb, feat[lvl + 1], B * np, ns, cout, Ctot, coff, total);
            }
            coff += cout;
        }
        Ccur[lvl + 1] = SA_DIMS[lvl][0][3] + SA_DIMS[lvl][1][3];
    }

    // ---- feature propagation ----
    for (int i = 4; i >= 1; --i) {
        int nq = NP[i - 1], nsrc = NP[i];
        {
            int total = B * nq;
            threenn_kernel<<<CDIV(total, TPB), TPB, 0, stream>>>(
                xyz[i - 1], xyz[i], idx3, w3, nq, nsrc, total);
        }
        int Cs = Ccur[i], Ck = Ccur[i - 1];
        int Kp = rup(Cs + Ck, 32);
        int rows = B * nq;
        {
            long long total = (long long)rows * Kp;
            interp_concat_kernel<<<CDIV(total, TPB), TPB, 0, stream>>>(
                feat[i], feat[i - 1], idx3, w3, scrA, nq, nsrc, Cs, Ck, Kp, total);
        }
        const float* in   = scrA;
        int          kdim = Cs + Ck;
        int          lda  = Kp;
        for (int L = 0; L < 2; ++L) {
            int    ndim = FP_DIMS[i - 1][L + 1];
            float* outb = (L == 0) ? scrB : feat[i - 1];
            lda = gemm(rows, in, lda, fpW[i - 1][L], kdim, ndim,
                       fpS[i - 1][L], fpB[i - 1][L], outb, 1);
            in = outb; kdim = ndim;
        }
        Ccur[i - 1] = FP_DIMS[i - 1][2];     // mult of 32 -> stride == C
    }

    // ---- head: fc1 (relu) -> fc2 -> transpose ----
    {
        int rows = B * N0;
        gemm(rows, feat[0], 128, fc1W, 128, 128, fc1S, fc1B, scrA, 1);
        int ldc2 = gemm(rows, scrA, 128, fc2W, 128, 13, nullptr, fc2B, scrB, 0);
        int total = B * 13 * N0;
        transpose_out_kernel<<<CDIV(total, TPB), TPB, 0, stream>>>(
            scrB, (float*)d_out, N0, 13, ldc2, total);
    }
}
